// MyLSTM_34076270527171
// MI455X (gfx1250) — compile-verified, matrix operand order checked by reading, not run on hardware
//
#include <hip/hip_runtime.h>
#include <hip/hip_bf16.h>

typedef __attribute__((ext_vector_type(16))) __bf16 v16bf;
typedef __attribute__((ext_vector_type(8)))  __bf16 bf16x8;
typedef __attribute__((ext_vector_type(8)))  float  v8f;

#define TSEQ 1024
#define BM   16
#define K1   192   // 64 (x) + 128 (h1)
#define N1   512   // 4*H
#define K2   192   // 128 (h1) + 64 (h2)
#define N2   256   // 4*H2

// sigmoid via hardware v_exp_f32 (TRANS, co-executes with WMMA)
static __device__ __forceinline__ float sigmoidf_(float x) {
    return 1.0f / (1.0f + __expf(-x));
}
// tanh via hardware exp: 1 - 2/(e^{2x}+1); saturates to +/-1 for large |x|
static __device__ __forceinline__ float tanhf_(float x) {
    float e = __expf(2.0f * x);
    return 1.0f - 2.0f / (e + 1.0f);
}

union FragU { bf16x8 h[2]; v16bf v; };

// CDNA5 16-bit A/B fragment layout: lanes 0-15 hold row/col 0-15 with K pairs
// {kblk*32 + 0..7, kblk*32 + 16..23}; lanes 16-31 hold K {+8..15, +24..31}.
static __device__ __forceinline__ v16bf load_frag(const __bf16* base, int ldk,
                                                  int row, int kblk, int hi) {
    const __bf16* p = base + row * ldk + kblk * 32 + (hi << 3);
    FragU u;
    u.h[0] = *(const bf16x8*)(p);
    u.h[1] = *(const bf16x8*)(p + 16);
    return u.v;
}

__global__ void __launch_bounds__(256)
lstm_fused_kernel(const float* __restrict__ x,
                  const float* __restrict__ Wih1, const float* __restrict__ Whh1,
                  const float* __restrict__ bih1, const float* __restrict__ bhh1,
                  const float* __restrict__ Wih2, const float* __restrict__ Whh2,
                  const float* __restrict__ bih2, const float* __restrict__ bhh2,
                  const float* __restrict__ W1,   const float* __restrict__ b1,
                  const float* __restrict__ W2,   const float* __restrict__ b2,
                  float* __restrict__ out)
{
    extern __shared__ char smem_raw[];
    __bf16* Wc1  = (__bf16*)smem_raw;          // [N1][K1] bf16 = 192 KB
    __bf16* A1   = Wc1 + N1 * K1;              // [BM][K1]  x_t | h1
    __bf16* A2   = A1 + BM * K1;               // [BM][K2]  h1  | h2
    float* gates = (float*)(A2 + BM * K2);     // [BM][N1] (reused for layer 2)
    float* c1    = gates + BM * N1;            // [BM][128]
    float* c2    = c1 + BM * 128;              // [BM][64]
    float* h2f   = c2 + BM * 64;               // [BM][64]
    float* bias1 = h2f + BM * 64;              // [N1]
    float* bias2 = bias1 + N1;                 // [N2]

    const int tid  = threadIdx.x;
    const int lane = tid & 31;
    const int wave = tid >> 5;
    const int col  = lane & 15;
    const int hi   = lane >> 4;
    const int b0   = blockIdx.x * BM;

    // ---- one-time preload: layer-1 concat weights -> LDS (bf16, [n][k]) ----
    for (int idx = tid; idx < N1 * K1; idx += 256) {
        int n = idx / K1, k = idx - n * K1;
        float w = (k < 64) ? Wih1[n * 64 + k] : Whh1[n * 128 + (k - 64)];
        Wc1[idx] = (__bf16)w;
    }
    for (int i = tid; i < N1; i += 256) bias1[i] = bih1[i] + bhh1[i];
    for (int i = tid; i < N2; i += 256) bias2[i] = bih2[i] + bhh2[i];
    for (int i = tid; i < BM * 128; i += 256) c1[i] = 0.f;
    for (int i = tid; i < BM * 64;  i += 256) c2[i] = 0.f;
    for (int i = tid; i < BM * K1;  i += 256) A1[i] = (__bf16)0.f;
    for (int i = tid; i < BM * K2;  i += 256) A2[i] = (__bf16)0.f;

    // ---- one-time: layer-2 concat weight fragments -> registers ----
    v16bf Bw2[2][6];
#pragma unroll
    for (int nt = 0; nt < 2; ++nt) {
        const int n = (wave * 2 + nt) * 16 + col;
#pragma unroll
        for (int kf = 0; kf < 6; ++kf) {
            v16bf v;
#pragma unroll
            for (int e = 0; e < 16; ++e) {
                int k = kf * 32 + ((e < 8) ? (hi * 8 + e) : (16 + hi * 8 + (e - 8)));
                float w = (k < 128) ? Wih2[n * 128 + k] : Whh2[n * 64 + (k - 128)];
                v[e] = (__bf16)w;
            }
            Bw2[nt][kf] = v;
        }
    }

    // ---- x double-buffer setup: each thread owns 4 contiguous elements ----
    const int xr = (tid * 4) >> 6;             // batch row within tile
    const int xi = (tid * 4) & 63;             // input-feature offset
    const float* xbase = x + (size_t)(b0 + xr) * TSEQ * 64 + xi;
    __bf16* xdst = A1 + xr * K1 + xi;

    // x_0 -> LDS now; x_1 -> registers (consumed next step)
    {
        float4 v0 = *(const float4*)(xbase);
        xdst[0] = (__bf16)v0.x; xdst[1] = (__bf16)v0.y;
        xdst[2] = (__bf16)v0.z; xdst[3] = (__bf16)v0.w;
    }
    float4 xv = *(const float4*)(xbase + 64);
    __syncthreads();

    for (int t = 0; t < TSEQ; ++t) {
        // phase 1: GEMM1  [16,192] x [192 -> 512] gates, bf16 WMMA, f32 acc
        {
            v16bf a[6];
#pragma unroll
            for (int kf = 0; kf < 6; ++kf) a[kf] = load_frag(A1, K1, col, kf, hi);
#pragma unroll
            for (int nt = 0; nt < 4; ++nt) {
                const int ntile = wave * 4 + nt;
                v8f acc = {};
#pragma unroll
                for (int kf = 0; kf < 6; ++kf) {
                    v16bf bfr = load_frag(Wc1, K1, ntile * 16 + col, kf, hi);
                    acc = __builtin_amdgcn_wmma_f32_16x16x32_bf16(
                        false, a[kf], false, bfr, (short)0, acc, false, false);
                }
#pragma unroll
                for (int r = 0; r < 8; ++r)
                    gates[(r + hi * 8) * N1 + ntile * 16 + col] = acc[r];
            }
        }
        __syncthreads();

        // phase 2: LSTM cell 1 (16 x 128) + x double-buffer rotation
        //  - A1 x-region is dead after phase 1, so stage x_{t+1} here
        if (t + 1 < TSEQ) {
            xdst[0] = (__bf16)xv.x; xdst[1] = (__bf16)xv.y;
            xdst[2] = (__bf16)xv.z; xdst[3] = (__bf16)xv.w;
        }
        if (t + 2 < TSEQ)
            xv = *(const float4*)(xbase + (size_t)(t + 2) * 64);
#pragma unroll
        for (int p = 0; p < 8; ++p) {
            int idx = tid + p * 256;
            int r = idx >> 7, j = idx & 127;
            const float* g = gates + r * N1;
            float gi = sigmoidf_(g[j]       + bias1[j]);
            float gf = sigmoidf_(g[128 + j] + bias1[128 + j]);
            float gg = tanhf_   (g[256 + j] + bias1[256 + j]);
            float go = sigmoidf_(g[384 + j] + bias1[384 + j]);
            float c  = gf * c1[idx] + gi * gg;
            c1[idx] = c;
            float h = go * tanhf_(c);
            __bf16 hb = (__bf16)h;
            A1[r * K1 + 64 + j] = hb;
            A2[r * K2 + j]      = hb;
        }
        __syncthreads();

        // phase 3: GEMM2  [16,192] x [192 -> 256], B fragments from registers
        {
            v16bf a[6];
#pragma unroll
            for (int kf = 0; kf < 6; ++kf) a[kf] = load_frag(A2, K2, col, kf, hi);
#pragma unroll
            for (int nt = 0; nt < 2; ++nt) {
                const int ntile = wave * 2 + nt;
                v8f acc = {};
#pragma unroll
                for (int kf = 0; kf < 6; ++kf)
                    acc = __builtin_amdgcn_wmma_f32_16x16x32_bf16(
                        false, a[kf], false, Bw2[nt][kf], (short)0, acc, false, false);
#pragma unroll
                for (int r = 0; r < 8; ++r)
                    gates[(r + hi * 8) * N2 + ntile * 16 + col] = acc[r];
            }
        }
        __syncthreads();

        // phase 4: LSTM cell 2 (16 x 64), write h2 into A2 and h2f
#pragma unroll
        for (int p = 0; p < 4; ++p) {
            int idx = tid + p * 256;
            int r = idx >> 6, j = idx & 63;
            const float* g = gates + r * N2;
            float gi = sigmoidf_(g[j]       + bias2[j]);
            float gf = sigmoidf_(g[64 + j]  + bias2[64 + j]);
            float gg = tanhf_   (g[128 + j] + bias2[128 + j]);
            float go = sigmoidf_(g[192 + j] + bias2[192 + j]);
            float c  = gf * c2[idx] + gi * gg;
            c2[idx] = c;
            float h = go * tanhf_(c);
            A2[r * K2 + 128 + j] = (__bf16)h;
            h2f[idx] = h;
        }
        __syncthreads();   // protects: gates (next phase 1 writes), h2f (head)
    }

    // ---- MLP head: sigmoid(sigmoid(h2 @ W1^T + b1) @ W2^T + b2) ----
    if (tid < BM) {
        float acc2 = b2[0];
        for (int jj = 0; jj < 32; ++jj) {
            float s = b1[jj];
            const float* w = W1 + jj * 64;
            for (int k = 0; k < 64; ++k) s += h2f[tid * 64 + k] * w[k];
            acc2 += sigmoidf_(s) * W2[jj];
        }
        out[b0 + tid] = sigmoidf_(acc2);
    }
}

extern "C" void kernel_launch(void* const* d_in, const int* in_sizes, int n_in,
                              void* d_out, int out_size, void* d_ws, size_t ws_size,
                              hipStream_t stream) {
    const float* x    = (const float*)d_in[0];
    const float* Wih1 = (const float*)d_in[1];
    const float* Whh1 = (const float*)d_in[2];
    const float* bih1 = (const float*)d_in[3];
    const float* bhh1 = (const float*)d_in[4];
    const float* Wih2 = (const float*)d_in[5];
    const float* Whh2 = (const float*)d_in[6];
    const float* bih2 = (const float*)d_in[7];
    const float* bhh2 = (const float*)d_in[8];
    const float* W1   = (const float*)d_in[9];
    const float* b1   = (const float*)d_in[10];
    const float* W2   = (const float*)d_in[11];
    const float* b2   = (const float*)d_in[12];
    float* out = (float*)d_out;

    const size_t smem =
        (size_t)(N1 * K1 + BM * K1 + BM * K2) * sizeof(__bf16) +   // bf16 region
        (size_t)(BM * N1 + BM * 128 + BM * 64 + BM * 64 + N1 + N2) * sizeof(float);

    (void)hipFuncSetAttribute((const void*)lstm_fused_kernel,
                              hipFuncAttributeMaxDynamicSharedMemorySize,
                              (int)smem);

    lstm_fused_kernel<<<dim3(256 / BM), dim3(256), smem, stream>>>(
        x, Wih1, Whh1, bih1, bhh1, Wih2, Whh2, bih2, bhh2, W1, b1, W2, b2, out);
}
